// POTAlignment_56573309223310
// MI455X (gfx1250) — matrix operand order repaired and checked by reading, not compiled, not run on hardware
//
#include <hip/hip_runtime.h>
#include <hip/hip_bf16.h>
#include <stdint.h>

// ---------------------------------------------------------------------------
// POT Sinkhorn on MI455X (gfx1250, wave32)
//   Phase 0: one-shot f32 -> bf16(hi,lo) split of S and T (kills per-tile cvt)
//   Phase 1: row norms
//   Phase 2: WMMA GEMM (bf16x3) -> C, K   (128 MB, L2-resident)
//   Phase 3: 100 Sinkhorn iterations; vectors staged in LDS via
//            global_load_async_to_lds_b128 (ASYNCcnt path)
//   Phase 4: deterministic two-stage loss reduction
// ---------------------------------------------------------------------------

typedef __attribute__((ext_vector_type(16))) __bf16 v16bf;
typedef __attribute__((ext_vector_type(8)))  __bf16 v8bf;
typedef __attribute__((ext_vector_type(8)))  float  v8f;

#define NPTS  4096
#define DIM   512
#define INV_N (1.0f / 4096.0f)
#define EPS_S 1e-8f
// K = exp(-C / 0.1) => exp(-10*C)

__device__ __forceinline__ float wave_reduce(float v) {
#pragma unroll
  for (int off = 16; off > 0; off >>= 1) v += __shfl_xor(v, off, 32);
  return v;
}

// CDNA5 async global -> LDS copy (16 bytes per lane), tracked by ASYNCcnt.
__device__ __forceinline__ void async_copy_b128(uint32_t lds_off, const void* gptr) {
  asm volatile("global_load_async_to_lds_b128 %0, %1, off"
               :: "v"(lds_off), "v"((uint64_t)(uintptr_t)gptr) : "memory");
}
__device__ __forceinline__ void wait_async0() {
  asm volatile("s_wait_asynccnt 0x0" ::: "memory");
}

// ---------------- Phase 0: bf16 hi/lo split ---------------------------------
__global__ void pot_split_kernel(const float* __restrict__ S,
                                 const float* __restrict__ T,
                                 __bf16* __restrict__ Shi, __bf16* __restrict__ Slo,
                                 __bf16* __restrict__ Thi, __bf16* __restrict__ Tlo) {
  int i = blockIdx.x * blockDim.x + threadIdx.x;   // 0 .. 2M-1
  float x = S[i];
  __bf16 h = (__bf16)x;
  Shi[i] = h;
  Slo[i] = (__bf16)(x - (float)h);
  float y = T[i];
  h = (__bf16)y;
  Thi[i] = h;
  Tlo[i] = (__bf16)(y - (float)h);
}

// ---------------- Phase 1: squared row norms --------------------------------
__global__ void pot_norms_kernel(const float* __restrict__ S,
                                 const float* __restrict__ T,
                                 float* __restrict__ x2, float* __restrict__ y2) {
  int gw   = (int)((blockIdx.x * blockDim.x + threadIdx.x) >> 5);  // wave id = row
  int lane = (int)(threadIdx.x & 31);
  const float* p = (gw < NPTS) ? (S + (size_t)gw * DIM)
                               : (T + (size_t)(gw - NPTS) * DIM);
  const float4* p4 = (const float4*)(p + lane * 16);   // 16 floats per lane
  float s = 0.f;
#pragma unroll
  for (int t = 0; t < 4; ++t) {
    float4 q = p4[t];
    s += q.x * q.x + q.y * q.y + q.z * q.z + q.w * q.w;
  }
  s = wave_reduce(s);
  if (lane == 0) {
    if (gw < NPTS) x2[gw] = s; else y2[gw - NPTS] = s;
  }
}

// ---------------- Phase 2: WMMA GEMM + exp ----------------------------------
// One wave per 16x16 tile. Inner loop: pure bf16 loads + 3 WMMAs.
//   G ~= Ahi*Bhi + Ahi*Blo + Alo*Bhi   (f32 accumulate)
// A frag (16x32 bf16): lane holds row m = lane%16, k = hf*16 + e (contiguous)
// B frag (32x16 bf16): lane holds col n = lane%16, k = hf*16 + e (contiguous)
__global__ void pot_gemm_exp_kernel(const __bf16* __restrict__ Shi,
                                    const __bf16* __restrict__ Slo,
                                    const __bf16* __restrict__ Thi,
                                    const __bf16* __restrict__ Tlo,
                                    const float* __restrict__ x2,
                                    const float* __restrict__ y2,
                                    float* __restrict__ Kmat,
                                    float* __restrict__ Cmat) {
  const int lane = (int)(threadIdx.x & 31);
  const int wv   = (int)(threadIdx.x >> 5);
  const int tile_n = (blockIdx.x * 4 + wv) * 16;
  const int tile_m = blockIdx.y * 16;
  const int r  = lane & 15;   // row (A) / column (B) within tile
  const int hf = lane >> 4;   // lane half selects k sub-chunk

  const __bf16* ah = Shi + (size_t)(tile_m + r) * DIM;
  const __bf16* al = Slo + (size_t)(tile_m + r) * DIM;
  const __bf16* bh = Thi + (size_t)(tile_n + r) * DIM;
  const __bf16* bl = Tlo + (size_t)(tile_n + r) * DIM;

  v8f acc = {};

  for (int kk = 0; kk < DIM; kk += 32) {
    if (kk + 32 < DIM) {
      __builtin_prefetch(ah + kk + 32, 0, 3);
      __builtin_prefetch(bh + kk + 32, 0, 3);
    }
    // A fragments: 16 contiguous bf16 (32B)
    v16bf a_hi = *(const v16bf*)(ah + kk + hf * 16);
    v16bf a_lo = *(const v16bf*)(al + kk + hf * 16);
    // B fragments: 16 contiguous bf16 (32B)
    v16bf b_hi = *(const v16bf*)(bh + kk + hf * 16);
    v16bf b_lo = *(const v16bf*)(bl + kk + hf * 16);

    acc = __builtin_amdgcn_wmma_f32_16x16x32_bf16(false, a_hi, false, b_hi,
                                                  (short)0, acc, false, false);
    acc = __builtin_amdgcn_wmma_f32_16x16x32_bf16(false, a_hi, false, b_lo,
                                                  (short)0, acc, false, false);
    acc = __builtin_amdgcn_wmma_f32_16x16x32_bf16(false, a_lo, false, b_hi,
                                                  (short)0, acc, false, false);
  }

  // C/D layout: VGPR v -> row (v + 8*hf), col = lane%16
  const int   col = tile_n + r;
  const float yv  = y2[col];
#pragma unroll
  for (int v = 0; v < 8; ++v) {
    int   row = tile_m + v + hf * 8;
    float g   = acc[v];
    float c   = x2[row] + yv - 2.0f * g;
    size_t idx = (size_t)row * NPTS + col;
    Cmat[idx] = c;
    Kmat[idx] = __expf(-10.0f * c);   // exp(-C/EPSILON), EPSILON=0.1
  }
}

// ---------------- Phase 3: Sinkhorn iterations ------------------------------
__global__ void pot_init_kernel(float* __restrict__ b, float* __restrict__ out) {
  int i = blockIdx.x * blockDim.x + threadIdx.x;
  if (i < NPTS) b[i] = 0.5f;            // (n * PORTION) / m = 0.5
  if (i == 0) out[0] = 0.f;
}

// u[row] = (1/n) / (K[row,:] . b + eps)  — one wave per row.
// b is async-staged into LDS once per block (16 KB), dots read via ds_load.
__global__ void pot_u_kernel(const float* __restrict__ Kmat,
                             const float* __restrict__ b,
                             float* __restrict__ u) {
  __shared__ float bsh[NPTS];
  uint32_t lds_base = (uint32_t)(size_t)&bsh[0];
#pragma unroll
  for (int t = 0; t < 4; ++t) {
    uint32_t off16 = (uint32_t)(t * 256 + threadIdx.x) * 16u;  // bytes
    async_copy_b128(lds_base + off16, (const char*)b + off16);
  }
  wait_async0();
  __syncthreads();

  int row  = (int)((blockIdx.x * blockDim.x + threadIdx.x) >> 5);
  int lane = (int)(threadIdx.x & 31);
  const float4* kr = (const float4*)(Kmat + (size_t)row * NPTS);
  const float4* b4 = (const float4*)bsh;
  float s = 0.f;
#pragma unroll 4
  for (int j = lane; j < NPTS / 4; j += 32) {
    float4 k = kr[j], v = b4[j];
    s += k.x * v.x + k.y * v.y + k.z * v.z + k.w * v.w;
  }
  s = wave_reduce(s);
  if (lane == 0) u[row] = INV_N / (s + EPS_S);
}

// b[col] /= (K[:,col] . u + eps)  — thread per column, coalesced row sweeps.
// u async-staged into LDS once per block.
__global__ void pot_b_kernel(const float* __restrict__ Kmat,
                             const float* __restrict__ u,
                             float* __restrict__ b) {
  __shared__ float ush[NPTS];
  uint32_t lds_base = (uint32_t)(size_t)&ush[0];
#pragma unroll
  for (int t = 0; t < 4; ++t) {
    uint32_t off16 = (uint32_t)(t * 256 + threadIdx.x) * 16u;  // bytes
    async_copy_b128(lds_base + off16, (const char*)u + off16);
  }
  wait_async0();
  __syncthreads();

  int col = blockIdx.x * blockDim.x + threadIdx.x;
  float s = 0.f;
#pragma unroll 8
  for (int i = 0; i < NPTS; ++i) s += Kmat[(size_t)i * NPTS + col] * ush[i];
  b[col] = b[col] / (s + EPS_S);
}

// ---------------- Phase 4: loss = sum a_i*K_ij*b_j*C_ij ---------------------
__global__ void pot_loss_partial_kernel(const float* __restrict__ Kmat,
                                        const float* __restrict__ Cmat,
                                        const float* __restrict__ b,
                                        float* __restrict__ partial) {
  __shared__ float sm[8];
  size_t base = (size_t)blockIdx.x * blockDim.x * 16;
  float s = 0.f;
#pragma unroll 4
  for (int t = 0; t < 16; ++t) {
    size_t idx = base + (size_t)t * blockDim.x + threadIdx.x;
    int j = (int)(idx & (NPTS - 1));
    s += Kmat[idx] * Cmat[idx] * b[j];
  }
  s = wave_reduce(s);
  int lane = (int)(threadIdx.x & 31), wv = (int)(threadIdx.x >> 5);
  if (lane == 0) sm[wv] = s;
  __syncthreads();
  if (wv == 0) {
    float v = (lane < 8) ? sm[lane] : 0.f;
    v = wave_reduce(v);
    if (lane == 0) partial[blockIdx.x] = v;
  }
}

__global__ void pot_loss_final_kernel(const float* __restrict__ partial,
                                      float* __restrict__ out) {
  __shared__ float sm[256];
  float s = 0.f;
#pragma unroll
  for (int t = 0; t < 16; ++t) s += partial[t * 256 + threadIdx.x];
  sm[threadIdx.x] = s;
  __syncthreads();
#pragma unroll
  for (int off = 128; off > 0; off >>= 1) {
    if ((int)threadIdx.x < off) sm[threadIdx.x] += sm[threadIdx.x + off];
    __syncthreads();
  }
  if (threadIdx.x == 0) out[0] = sm[0] * INV_N;   // a_i = 1/n
}

// ---------------------------------------------------------------------------
extern "C" void kernel_launch(void* const* d_in, const int* in_sizes, int n_in,
                              void* d_out, int out_size, void* d_ws, size_t ws_size,
                              hipStream_t stream) {
  const float* S = (const float*)d_in[0];   // source [4096,512] f32
  const float* T = (const float*)d_in[1];   // target [4096,512] f32
  float* out = (float*)d_out;               // scalar loss

  // workspace layout (floats):
  //   K[16M] | C[16M] | x2[4K] | y2[4K] | b[4K] | u[4K] | partial[4K] | bf16 splits
  float* ws      = (float*)d_ws;
  float* Kmat    = ws;
  float* Cmat    = Kmat + (size_t)NPTS * NPTS;
  float* x2      = Cmat + (size_t)NPTS * NPTS;
  float* y2      = x2 + NPTS;
  float* bv      = y2 + NPTS;
  float* uv      = bv + NPTS;
  float* partial = uv + NPTS;
  __bf16* Shi    = (__bf16*)(partial + NPTS);
  __bf16* Slo    = Shi + (size_t)NPTS * DIM;
  __bf16* Thi    = Slo + (size_t)NPTS * DIM;
  __bf16* Tlo    = Thi + (size_t)NPTS * DIM;

  // Phase 0: split (2M elements per matrix)
  pot_split_kernel<<<8192, 256, 0, stream>>>(S, T, Shi, Slo, Thi, Tlo);

  // Phase 1: norms (8192 waves, 8 waves/block)
  pot_norms_kernel<<<1024, 256, 0, stream>>>(S, T, x2, y2);

  // Phase 2: WMMA GEMM + exp. 256x256 tiles, 4 waves/block along n.
  pot_gemm_exp_kernel<<<dim3(64, 256), 128, 0, stream>>>(Shi, Slo, Thi, Tlo,
                                                         x2, y2, Kmat, Cmat);

  // Phase 3: Sinkhorn
  pot_init_kernel<<<16, 256, 0, stream>>>(bv, out);
  for (int it = 0; it < 100; ++it) {
    pot_u_kernel<<<512, 256, 0, stream>>>(Kmat, bv, uv);   // 4096 waves
    pot_b_kernel<<<16, 256, 0, stream>>>(Kmat, uv, bv);    // 4096 threads
  }

  // Phase 4: loss (4096 blocks x 256 threads x 16 elems = 16M)
  pot_loss_partial_kernel<<<4096, 256, 0, stream>>>(Kmat, Cmat, bv, partial);
  pot_loss_final_kernel<<<1, 256, 0, stream>>>(partial, out);
}
